// PolicyGNN_36266703847448
// MI455X (gfx1250) — matrix operand order repaired and checked by reading, not compiled, not run on hardware
//
#include <hip/hip_runtime.h>

// ---------------------------------------------------------------------------
// PolicyGNN for MI455X (gfx1250): bf16 WMMA pipeline.
// Dominant cost: streaming mat (64 MB f32) once => HBM-bound (~2.8us @23.3TB/s).
// Weights are pre-packed once into bf16 WMMA B-fragment order so the hot loops
// do zero conversions and load each fragment as 2x global_load_b128.
// tmp1 (16MB), weights, and the 0.5GB gather all live in the 192MB L2.
// ---------------------------------------------------------------------------

typedef __attribute__((ext_vector_type(16))) __bf16    v16bf;
typedef __attribute__((ext_vector_type(4)))  __bf16    v4bf;
typedef __attribute__((ext_vector_type(8)))  float     v8f;
typedef __attribute__((ext_vector_type(8)))  unsigned  v8u;

#define NODES 4096
#define BATCH 8
#define DEG   32
#define MID   128

// f32 -> bf16, round-to-nearest-even (explicit bit math; no cast dependence)
__device__ __forceinline__ __bf16 f2bf(float x) {
    unsigned u = __builtin_bit_cast(unsigned, x);
    unsigned r = u + 0x7FFFu + ((u >> 16) & 1u);
    unsigned short h = (unsigned short)(r >> 16);
    return __builtin_bit_cast(__bf16, h);
}

__device__ __forceinline__ v8f wmma_bf16(v16bf a, v16bf b, v8f c) {
    // (neg_a, A, neg_b, B, c_mod, C, reuse_a, reuse_b)
    return __builtin_amdgcn_wmma_f32_16x16x32_bf16(false, a, false, b,
                                                   (short)0, c, false, false);
}

// A-fragment K index inside a 16x32 bf16 tile (matches B packing):
// lanes 0-15 hold K = {0..15}, lanes 16-31 hold K = {16..31}
__device__ __forceinline__ int a_kidx(int half, int i) {
    return half * 16 + i;
}

// Load one pre-packed 16-element bf16 B fragment (32 bytes -> 2x b128)
__device__ __forceinline__ v16bf load_bfrag(const __bf16* __restrict__ bq,
                                            int step, int half, int ncol) {
    const int off = (((step * 2 + half) * MID + ncol)) << 4;
    return __builtin_bit_cast(v16bf, *(const v8u*)&bq[off]);
}

// ---------------------------------------------------------------------------
// Weight pre-pack: w [Ktot x 128] f32 (rows >= Kvalid are zero) -> bf16
// fragment order bq[step][half][ncol][i], i contiguous (16 elems = 32B).
// ---------------------------------------------------------------------------
__global__ __launch_bounds__(256) void k_prep_w(
    const float* __restrict__ w, __bf16* __restrict__ bq,
    int Kvalid, int Ktot)
{
    const int idx = blockIdx.x * 256 + threadIdx.x;
    const int total = (Ktot / 32) * 2 * MID * 16;
    if (idx >= total) return;
    const int i    = idx & 15;
    const int ncol = (idx >> 4) & 127;
    const int half = (idx >> 11) & 1;
    const int step = idx >> 12;
    const int k = step * 32 + half * 16 + i;
    bq[idx] = f2bf(k < Kvalid ? w[k * MID + ncol] : 0.f);
}

// ---------------------------------------------------------------------------
// Kernel 1: mat_enc  m = relu(mat @ me_w1 + b1) @ me_w2 + b2   -> [4096, 8]
// One WG per 16-row strip; 8 waves each own a 16-wide slab of MID=128.
// K-step 64 (2 WMMA/iter), double-buffered LDS A tile, 1 barrier/iter.
// ---------------------------------------------------------------------------
__global__ __launch_bounds__(256) void k_mat_enc(
    const float* __restrict__ mat,    // [4096,4096]
    const __bf16* __restrict__ bq1,   // me_w1 pre-packed fragments
    const float* __restrict__ me_b1,  // [128]
    const float* __restrict__ me_w2,  // [128,8]
    const float* __restrict__ me_b2,  // [8]
    float* __restrict__ m_out)        // [4096,8]
{
    __shared__ __bf16 ldsA[2][16][64];
    __shared__ float  ldsMid[16][MID];

    const int node0 = blockIdx.x * 16;
    const int tid  = threadIdx.x;
    const int wave = tid >> 5;
    const int lane = tid & 31;
    const int half = lane >> 4;
    const int l16  = lane & 15;
    const int ncol = wave * 16 + l16;   // output column this lane produces
    const int lrow = tid >> 4;          // A-tile loader: row, 4 cols/thread
    const int lcol = (tid & 15) * 4;

    // prologue: stage tile 0
    {
        float4 v0 = *(const float4*)&mat[(size_t)(node0 + lrow) * NODES + lcol];
        v4bf p; p[0] = f2bf(v0.x); p[1] = f2bf(v0.y);
                p[2] = f2bf(v0.z); p[3] = f2bf(v0.w);
        *(v4bf*)&ldsA[0][lrow][lcol] = p;
    }
    __syncthreads();

    v8f c = {};
    for (int t = 0; t < NODES / 64; ++t) {
        const int cur = t & 1;
        const bool have = (t < NODES / 64 - 1);
        float4 nv;
        if (have)   // issue next tile's HBM loads before consuming current
            nv = *(const float4*)&mat[(size_t)(node0 + lrow) * NODES
                                      + (t + 1) * 64 + lcol];
        #pragma unroll
        for (int j = 0; j < 2; ++j) {
            v16bf a;
            #pragma unroll
            for (int i = 0; i < 16; ++i)
                a[i] = ldsA[cur][l16][j * 32 + a_kidx(half, i)];
            v16bf bm = load_bfrag(bq1, t * 2 + j, half, ncol);
            c = wmma_bf16(a, bm, c);
        }
        if (have) {
            v4bf p; p[0] = f2bf(nv.x); p[1] = f2bf(nv.y);
                    p[2] = f2bf(nv.z); p[3] = f2bf(nv.w);
            *(v4bf*)&ldsA[cur ^ 1][lrow][lcol] = p;
        }
        __syncthreads();
    }

    // bias + relu; C layout: VGPR r -> row (r + 8*half), col = ncol
    const float bias = me_b1[ncol];
    #pragma unroll
    for (int r = 0; r < 8; ++r) {
        float v = c[r] + bias;
        ldsMid[r + 8 * half][ncol] = v > 0.f ? v : 0.f;
    }
    __syncthreads();

    // layer 2: [16x128] @ [128x8] + b2 (tiny -> scalar, 1 output/thread)
    if (tid < 128) {
        int row = tid >> 3, col = tid & 7;
        float acc = me_b2[col];
        #pragma unroll 8
        for (int k = 0; k < MID; ++k)
            acc += ldsMid[row][k] * me_w2[k * 8 + col];
        m_out[(node0 + row) * 8 + col] = acc;
    }
}

// ---------------------------------------------------------------------------
// Kernel 2: inp_enc  tmp1 = relu(concat(inp, m) @ ie_w1 + b1) @ ie_w2 + b2
// One WG per (batch, 16-node) tile. K padded 40 -> 64 for stage 1.
// ---------------------------------------------------------------------------
__global__ __launch_bounds__(256) void k_inp_enc(
    const float* __restrict__ inp,    // [8,4096,32]
    const float* __restrict__ m_in,   // [4096,8]
    const __bf16* __restrict__ bq_w1, // ie_w1 packed (K 40->64, zero pad)
    const float* __restrict__ ie_b1,  // [128]
    const __bf16* __restrict__ bq_w2, // ie_w2 packed (K 128)
    const float* __restrict__ ie_b2,  // [128]
    float* __restrict__ tmp1)         // [8,4096,128]
{
    __shared__ __bf16 ldsA[16][64];
    __shared__ __bf16 ldsH[16][MID];

    const int tile  = blockIdx.x;
    const int b     = tile >> 8;            // 256 tiles per batch
    const int node0 = (tile & 255) * 16;
    const int tid  = threadIdx.x;
    const int wave = tid >> 5;
    const int lane = tid & 31;
    const int half = lane >> 4;
    const int l16  = lane & 15;
    const int ncol = wave * 16 + l16;

    // build concat A tile [16 x 64] (cols 0-31 inp, 32-39 m, 40-63 zero pad)
    for (int e = tid; e < 16 * 64; e += 256) {
        int r = e >> 6, cc = e & 63;
        float v;
        if (cc < 32)      v = inp[((size_t)b * NODES + node0 + r) * 32 + cc];
        else if (cc < 40) v = m_in[(node0 + r) * 8 + (cc - 32)];
        else              v = 0.f;
        ldsA[r][cc] = f2bf(v);
    }
    __syncthreads();

    // stage 1: K = 64 (2 WMMA)
    v8f c = {};
    #pragma unroll
    for (int s = 0; s < 2; ++s) {
        v16bf a;
        #pragma unroll
        for (int i = 0; i < 16; ++i)
            a[i] = ldsA[l16][s * 32 + a_kidx(half, i)];
        c = wmma_bf16(a, load_bfrag(bq_w1, s, half, ncol), c);
    }
    {
        const float bias = ie_b1[ncol];
        #pragma unroll
        for (int r = 0; r < 8; ++r) {
            float v = c[r] + bias;
            ldsH[r + 8 * half][ncol] = f2bf(v > 0.f ? v : 0.f);
        }
    }
    __syncthreads();

    // stage 2: K = 128 (4 WMMA)
    v8f c2 = {};
    #pragma unroll
    for (int s = 0; s < 4; ++s) {
        v16bf a;
        #pragma unroll
        for (int i = 0; i < 16; ++i)
            a[i] = ldsH[l16][s * 32 + a_kidx(half, i)];
        c2 = wmma_bf16(a, load_bfrag(bq_w2, s, half, ncol), c2);
    }
    {
        const float b2v = ie_b2[ncol];
        #pragma unroll
        for (int r = 0; r < 8; ++r)
            tmp1[((size_t)b * NODES + node0 + r + 8 * half) * MID + ncol] =
                c2[r] + b2v;
    }
}

// ---------------------------------------------------------------------------
// Kernel 3: neighbor gather-sum + te layers; only channel 0 of te_w2 matters
// (softmax over node axis is per-channel; reference keeps channel 0).
// ---------------------------------------------------------------------------
__global__ __launch_bounds__(256) void k_gather_te(
    const float* __restrict__ tmp1,   // [8,4096,128]
    const int*   __restrict__ ids,    // [4096,32]
    const __bf16* __restrict__ bq_w1, // te_w1 packed (K 256)
    const float* __restrict__ te_b1,  // [128]
    const float* __restrict__ te_w2,  // [128,2]
    const float* __restrict__ te_b2,  // [2]
    float* __restrict__ logits0)      // [8,4096]
{
    __shared__ __bf16 ldsA[16][256];
    __shared__ float  ldsH[16][MID];

    const int tile  = blockIdx.x;
    const int b     = tile >> 8;
    const int node0 = (tile & 255) * 16;
    const int tid  = threadIdx.x;
    const int wave = tid >> 5;
    const int lane = tid & 31;
    const int half = lane >> 4;
    const int l16  = lane & 15;
    const int ncol = wave * 16 + l16;

    // own tile -> A cols [0,128)
    for (int e = tid; e < 16 * MID; e += 256) {
        int r = e >> 7, cc = e & 127;
        ldsA[r][cc] = f2bf(tmp1[((size_t)b * NODES + node0 + r) * MID + cc]);
    }
    // gather-sum 32 neighbors (L2-resident) -> A cols [128,256)
    {
        int row = tid >> 4;
        int c0  = (tid & 15) * 8;
        float acc[8] = {0.f, 0.f, 0.f, 0.f, 0.f, 0.f, 0.f, 0.f};
        const int* idrow = &ids[(node0 + row) * DEG];
        for (int d = 0; d < DEG; ++d) {
            int j = idrow[d];
            const float* src = &tmp1[((size_t)b * NODES + j) * MID + c0];
            #pragma unroll
            for (int i = 0; i < 8; ++i) acc[i] += src[i];
        }
        #pragma unroll
        for (int i = 0; i < 8; ++i) ldsA[row][MID + c0 + i] = f2bf(acc[i]);
    }
    __syncthreads();

    // te layer 1: K = 256 (8 WMMA)
    v8f c = {};
    #pragma unroll
    for (int s = 0; s < 8; ++s) {
        v16bf a;
        #pragma unroll
        for (int i = 0; i < 16; ++i)
            a[i] = ldsA[l16][s * 32 + a_kidx(half, i)];
        c = wmma_bf16(a, load_bfrag(bq_w1, s, half, ncol), c);
    }
    {
        const float bias = te_b1[ncol];
        #pragma unroll
        for (int r = 0; r < 8; ++r) {
            float v = c[r] + bias;
            ldsH[r + 8 * half][ncol] = v > 0.f ? v : 0.f;
        }
    }
    __syncthreads();

    // te layer 2, channel 0 only: [16x128] . te_w2[:,0] + b2[0]
    if (tid < 16) {
        float acc = te_b2[0];
        #pragma unroll 8
        for (int k = 0; k < MID; ++k)
            acc += ldsH[tid][k] * te_w2[k * 2];
        logits0[(size_t)b * NODES + node0 + tid] = acc;
    }
}

// ---------------------------------------------------------------------------
// Kernel 4: softmax over the node axis (4096) per batch row.
// ---------------------------------------------------------------------------
__global__ __launch_bounds__(256) void k_softmax(
    const float* __restrict__ lg, float* __restrict__ out)
{
    __shared__ float red[256];
    const int b = blockIdx.x, tid = threadIdx.x;
    const float* x = lg + (size_t)b * NODES;

    float mx = -3.402823466e38f;
    for (int i = tid; i < NODES; i += 256) mx = fmaxf(mx, x[i]);
    red[tid] = mx;
    __syncthreads();
    for (int s = 128; s > 0; s >>= 1) {
        if (tid < s) red[tid] = fmaxf(red[tid], red[tid + s]);
        __syncthreads();
    }
    mx = red[0];
    __syncthreads();

    float sum = 0.f;
    for (int i = tid; i < NODES; i += 256) sum += __expf(x[i] - mx);
    red[tid] = sum;
    __syncthreads();
    for (int s = 128; s > 0; s >>= 1) {
        if (tid < s) red[tid] += red[tid + s];
        __syncthreads();
    }
    const float inv = 1.f / red[0];
    for (int i = tid; i < NODES; i += 256)
        out[(size_t)b * NODES + i] = __expf(x[i] - mx) * inv;
}

// ---------------------------------------------------------------------------
extern "C" void kernel_launch(void* const* d_in, const int* in_sizes, int n_in,
                              void* d_out, int out_size, void* d_ws, size_t ws_size,
                              hipStream_t stream) {
    const float* inp   = (const float*)d_in[0];
    const float* mat   = (const float*)d_in[1];
    const int*   ids   = (const int*)  d_in[2];
    const float* me_w1 = (const float*)d_in[3];
    const float* me_b1 = (const float*)d_in[4];
    const float* me_w2 = (const float*)d_in[5];
    const float* me_b2 = (const float*)d_in[6];
    const float* ie_w1 = (const float*)d_in[7];
    const float* ie_b1 = (const float*)d_in[8];
    const float* ie_w2 = (const float*)d_in[9];
    const float* ie_b2 = (const float*)d_in[10];
    const float* te_w1 = (const float*)d_in[11];
    const float* te_b1 = (const float*)d_in[12];
    const float* te_w2 = (const float*)d_in[13];
    const float* te_b2 = (const float*)d_in[14];
    float* out = (float*)d_out;

    // workspace layout (bytes):
    //   m[4096*8] f32          @ 0           (128 KB)
    //   tmp1[8*4096*128] f32   @ 131072      (16 MB)
    //   logits0[8*4096] f32    @ 16908288    (128 KB)
    //   bq_me1  bf16           @ 17039360    (1 MB)
    //   bq_ie1  bf16           @ 18087936    (16 KB)
    //   bq_ie2  bf16           @ 18104320    (32 KB)
    //   bq_te1  bf16           @ 18137088    (64 KB)
    char* ws = (char*)d_ws;
    float*  m_buf     = (float*)(ws);
    float*  tmp1_buf  = (float*)(ws + 131072);
    float*  logit_buf = (float*)(ws + 16908288);
    __bf16* bq_me1    = (__bf16*)(ws + 17039360);
    __bf16* bq_ie1    = (__bf16*)(ws + 18087936);
    __bf16* bq_ie2    = (__bf16*)(ws + 18104320);
    __bf16* bq_te1    = (__bf16*)(ws + 18137088);

    // pre-pack weights into bf16 B-fragment order (tiny, L2-resident)
    k_prep_w<<<(128 * 4096 + 255) / 256, 256, 0, stream>>>(me_w1, bq_me1, 4096, 4096);
    k_prep_w<<<(2   * 4096 + 255) / 256, 256, 0, stream>>>(ie_w1, bq_ie1,   40,   64);
    k_prep_w<<<(4   * 4096 + 255) / 256, 256, 0, stream>>>(ie_w2, bq_ie2,  128,  128);
    k_prep_w<<<(8   * 4096 + 255) / 256, 256, 0, stream>>>(te_w1, bq_te1,  256,  256);

    k_mat_enc  <<<NODES / 16,         256, 0, stream>>>(mat, bq_me1, me_b1,
                                                        me_w2, me_b2, m_buf);
    k_inp_enc  <<<BATCH * NODES / 16, 256, 0, stream>>>(inp, m_buf, bq_ie1, ie_b1,
                                                        bq_ie2, ie_b2, tmp1_buf);
    k_gather_te<<<BATCH * NODES / 16, 256, 0, stream>>>(tmp1_buf, ids, bq_te1, te_b1,
                                                        te_w2, te_b2, logit_buf);
    k_softmax  <<<BATCH,              256, 0, stream>>>(logit_buf, out);
}
